// MultitaskLearner_51256139711103
// MI455X (gfx1250) — matrix operand order repaired and checked by reading, not compiled, not run on hardware
//
#include <hip/hip_runtime.h>
#include <hip/hip_bf16.h>

typedef __attribute__((ext_vector_type(16))) _Float16 v16h;
typedef __attribute__((ext_vector_type(8)))  _Float16 v8h;
typedef __attribute__((ext_vector_type(8)))  float    v8f;

#define B_  32
#define H_  128
#define W_  128
#define HW_ (H_*W_)
#define C_  128
#define NP_ (B_*HW_)      // 524288 pixels total
#define SWG_ 512          // stats workgroups

static __device__ __forceinline__ v8f wmma_f16(v16h a, v16h b, v8f c) {
    // D = A(16x32 f16) * B(32x16 f16) + C(16x16 f32)
    return __builtin_amdgcn_wmma_f32_16x16x32_f16(
        /*neg_a=*/false, a, /*neg_b=*/false, b,
        /*c_mod=*/(short)0, c, /*reuse_a=*/false, /*reuse_b=*/false);
}

static __device__ __forceinline__ v16h concat8(v8h lo, v8h hi) {
    return __builtin_shufflevector(lo, hi, 0,1,2,3,4,5,6,7,8,9,10,11,12,13,14,15);
}

// ---------------------------------------------------------------------------
// Kernel 0: repack weights to f16, K-contiguous layouts for WMMA A-fragments.
//   w1r[co][k32]  : k = tap*2+ci for k<18 (tap = dy*3+dx), zero-padded to 32
//   w2r[tap][co][ci] : ci contiguous (128)
// ---------------------------------------------------------------------------
__global__ void prep_weights_kernel(const float* __restrict__ w1,
                                    const float* __restrict__ w2,
                                    _Float16* __restrict__ w1r,
                                    _Float16* __restrict__ w2r,
                                    float* __restrict__ zerobuf) { // psum/psq x2
    int i = blockIdx.x * 256 + threadIdx.x;
    if (i < 128 * 32) {
        int k = i & 31, co = i >> 5;
        float v = 0.f;
        if (k < 18) {
            int tap = k >> 1, ci = k & 1;
            int dy = tap / 3, dx = tap % 3;
            v = w1[((co * 2 + ci) * 3 + dy) * 3 + dx];
        }
        w1r[i] = (_Float16)v;
    }
    if (i < 9 * 128 * 128) {
        int ci = i & 127, co = (i >> 7) & 127, tap = i >> 14;
        int dy = tap / 3, dx = tap % 3;
        w2r[i] = (_Float16)w2[((co * 128 + ci) * 3 + dy) * 3 + dx];
    }
    if (i < 4 * SWG_ * C_) zerobuf[i] = 0.f;   // not strictly needed; hygiene
}

// ---------------------------------------------------------------------------
// Kernel 1: conv1 (2 -> 128 ch, 3x3, pad 1) as single-K32 WMMA implicit GEMM.
// Output y1 = conv + b1, stored NHWC f16 (pre-BN).
// Grid: B*H*(W/16) blocks of 256 threads (8 waves; wave w -> co 16w..16w+15).
// ---------------------------------------------------------------------------
__global__ void conv1_kernel(const float* __restrict__ outp,   // [B,5,H,W] f32
                             const _Float16* __restrict__ w1r, // [128][32]
                             const float* __restrict__ b1,
                             _Float16* __restrict__ y1) {      // [B,H,W,128]
    __shared__ __align__(16) _Float16 xs[3 * 18 * 2];          // [row][col][ci]
    int wg = blockIdx.x;
    int xt = wg & 7;
    int yy = (wg >> 3) & 127;
    int b  = wg >> 10;
    int x0 = xt * 16;
    int tid = threadIdx.x;

    if (tid < 108) {                      // 3*18*2 halo tile of gj=output[:,0:2]
        int ci = tid & 1;
        int c  = (tid >> 1) % 18;
        int r  = (tid >> 1) / 18;
        int gy = yy + r - 1, gx = x0 + c - 1;
        float v = 0.f;
        if (gy >= 0 && gy < H_ && gx >= 0 && gx < W_)
            v = outp[((b * 5 + ci) * H_ + gy) * W_ + gx];
        xs[(r * 18 + c) * 2 + ci] = (_Float16)v;
    }
    __syncthreads();

    int lane = tid & 31, wv = tid >> 5;
    int hi = lane >> 4, n = lane & 15;
    int cob = wv * 16;

    // B fragment (contiguous K, matches A): lane half holds contiguous K = hi*16 + i
    v16h bf;
#pragma unroll
    for (int i = 0; i < 16; ++i) {
        int k = hi * 16 + i;
        float v = 0.f;
        if (k < 18) {
            int tap = k >> 1, ci = k & 1;
            int dy = tap / 3, dx = tap % 3;
            v = (float)xs[(dy * 18 + (n + dx)) * 2 + ci];
        }
        bf[i] = (_Float16)v;
    }
    // A fragment: halfs 0..15 = contiguous K hi*16..hi*16+15 (matches B)
    int co = cob + n;
    const v8h* wp = (const v8h*)w1r;                 // 4 chunks per co
    v16h af = concat8(wp[co * 4 + 2 * hi], wp[co * 4 + 2 * hi + 1]);

    v8f acc = {};
    acc = wmma_f16(af, bf, acc);

    // D: VGPR r -> (co = cob + hi*8 + r, pixel = x0 + n); 8 contiguous halfs
    int pix = (b * H_ + yy) * W_ + (x0 + n);
    v8h o;
#pragma unroll
    for (int r = 0; r < 8; ++r) {
        int cor = cob + hi * 8 + r;
        o[r] = (_Float16)(acc[r] + b1[cor]);
    }
    *(v8h*)(y1 + (size_t)pix * C_ + cob + hi * 8) = o;
}

// ---------------------------------------------------------------------------
// Kernel 2a: deterministic two-stage BN stats, stage 1 (coalesced partials).
// Thread (cg = tid&15, pl = tid>>4) streams v8h chunks; per-WG partials to ws.
// ---------------------------------------------------------------------------
__global__ void bn_partial_kernel(const _Float16* __restrict__ y,  // NHWC
                                  float* __restrict__ psum,        // [SWG_][128]
                                  float* __restrict__ psq) {       // [SWG_][128]
    int wg = blockIdx.x, tid = threadIdx.x;
    int cg = tid & 15, pl = tid >> 4;
    float s[8] = {0.f}, q[8] = {0.f};
    int base = wg * (NP_ / SWG_);                 // 1024 pixels per WG
    for (int it = 0; it < (NP_ / SWG_) / 16; ++it) {
        int pix = base + it * 16 + pl;
        v8h v = *(const v8h*)(y + (size_t)pix * C_ + cg * 8);
#pragma unroll
        for (int i = 0; i < 8; ++i) { float f = (float)v[i]; s[i] += f; q[i] += f * f; }
    }
    __shared__ float red[16][C_];
#pragma unroll
    for (int i = 0; i < 8; ++i) red[pl][cg * 8 + i] = s[i];
    __syncthreads();
    if (tid < C_) {
        float t = 0.f;
        for (int p = 0; p < 16; ++p) t += red[p][tid];
        psum[wg * C_ + tid] = t;
    }
    __syncthreads();
#pragma unroll
    for (int i = 0; i < 8; ++i) red[pl][cg * 8 + i] = q[i];
    __syncthreads();
    if (tid < C_) {
        float t = 0.f;
        for (int p = 0; p < 16; ++p) t += red[p][tid];
        psq[wg * C_ + tid] = t;
    }
}

// ---------------------------------------------------------------------------
// Kernel 2b: stage 2 — fixed-order tree reduction, fold BN into scale/shift.
// ---------------------------------------------------------------------------
__global__ void bn_finalize_kernel(const float* __restrict__ psum,
                                   const float* __restrict__ psq,
                                   const float* __restrict__ gamma,
                                   const float* __restrict__ beta,
                                   float* __restrict__ scale,
                                   float* __restrict__ shift) {
    int c = blockIdx.x, tid = threadIdx.x;        // 128 blocks x 256 threads
    __shared__ float ls[256], lq[256];
    float s = 0.f, q = 0.f;
    for (int w = tid; w < SWG_; w += 256) { s += psum[w * C_ + c]; q += psq[w * C_ + c]; }
    ls[tid] = s; lq[tid] = q;
    __syncthreads();
    for (int st = 128; st > 0; st >>= 1) {
        if (tid < st) { ls[tid] += ls[tid + st]; lq[tid] += lq[tid + st]; }
        __syncthreads();
    }
    if (tid == 0) {
        float mean = ls[0] / (float)NP_;
        float var  = lq[0] / (float)NP_ - mean * mean;
        float sc   = gamma[c] * rsqrtf(var + 1e-5f);
        scale[c] = sc;
        shift[c] = beta[c] - mean * sc;
    }
}

// ---------------------------------------------------------------------------
// Kernel 3: conv2 (128 -> 128 ch, 3x3, pad 1) implicit GEMM.
// 32-pixel N-tile, two 16x16 accumulators per wave (A-fragment reuse),
// 72 WMMAs / wave. BN+ReLU of layer 1 fused into the LDS staging.
// Grid: B*H*(W/32) blocks of 256 threads.
// ---------------------------------------------------------------------------
__global__ void conv2_kernel(const _Float16* __restrict__ y1,   // NHWC (pre-BN)
                             const float* __restrict__ scale,
                             const float* __restrict__ shift,
                             const _Float16* __restrict__ w2r,  // [9][128][128]
                             const float* __restrict__ b2,
                             _Float16* __restrict__ y2) {       // NHWC
    __shared__ __align__(32) _Float16 xs[3 * 34 * C_];          // 26112 B
    __shared__ float s_sc[C_], s_sh[C_];
    int wg = blockIdx.x;
    int xt = wg & 3;
    int yy = (wg >> 2) & 127;
    int b  = wg >> 9;
    int x0 = xt * 32;
    int tid = threadIdx.x;

    if (tid < C_) { s_sc[tid] = scale[tid]; s_sh[tid] = shift[tid]; }
    __syncthreads();

    for (int e = tid; e < 3 * 34 * C_; e += 256) {
        int ci = e & 127;
        int c  = (e >> 7) % 34;
        int r  = (e >> 7) / 34;
        int gy = yy + r - 1, gx = x0 + c - 1;
        float h = 0.f;
        if (gy >= 0 && gy < H_ && gx >= 0 && gx < W_) {
            float v = (float)y1[((size_t)(b * H_ + gy) * W_ + gx) * C_ + ci];
            h = fmaxf(0.f, v * s_sc[ci] + s_sh[ci]);   // BN + ReLU fused here
        }
        xs[e] = (_Float16)h;
    }
    __syncthreads();

    int lane = tid & 31, wv = tid >> 5;
    int hi = lane >> 4, n = lane & 15;
    int cob = wv * 16;
    int co = cob + n;

    const v8h* w2p = (const v8h*)w2r;   // [tap][co][16 chunks of 8 halfs]
    v8f acc0 = {}, acc1 = {};
#pragma unroll
    for (int tap = 0; tap < 9; ++tap) {
        int dy = tap / 3, dx = tap % 3;
        const _Float16* xrow = xs + (size_t)(dy * 34 + (n + dx)) * C_;
#pragma unroll
        for (int kb = 0; kb < 4; ++kb) {
            // A fragment (contiguous K per lane half, matches B), reused 2x
            const v8h* ap = w2p + ((size_t)(tap * 128 + co) * 16) + kb * 4 + hi * 2;
            v16h af = concat8(ap[0], ap[1]);
            // B fragments (contiguous K per lane half, matches A):
            // one aligned 32-byte run -> two ds_load_b128 each
            int koff = kb * 32 + hi * 16;
            v16h bf0 = *(const v16h*)(xrow + koff);           // pixels x0+n
            v16h bf1 = *(const v16h*)(xrow + 16 * C_ + koff); // pixels x0+16+n
            acc0 = wmma_f16(af, bf0, acc0);
            acc1 = wmma_f16(af, bf1, acc1);
        }
    }

    float bsum[8];
#pragma unroll
    for (int r = 0; r < 8; ++r) bsum[r] = b2[cob + hi * 8 + r];

    int pix0 = (b * H_ + yy) * W_ + (x0 + n);
    v8h o0, o1;
#pragma unroll
    for (int r = 0; r < 8; ++r) {
        o0[r] = (_Float16)(acc0[r] + bsum[r]);
        o1[r] = (_Float16)(acc1[r] + bsum[r]);
    }
    *(v8h*)(y2 + (size_t)pix0 * C_ + cob + hi * 8) = o0;
    *(v8h*)(y2 + (size_t)(pix0 + 16) * C_ + cob + hi * 8) = o1;
}

// ---------------------------------------------------------------------------
// Kernel 4: conv3 (1x1, 128 -> 2) with fused BN+ReLU on input; writes
// jmap_origin [B,2,H,W] f32. Vectorized v8h input reads.
// ---------------------------------------------------------------------------
__global__ void conv3_kernel(const _Float16* __restrict__ y2,
                             const float* __restrict__ scale,
                             const float* __restrict__ shift,
                             const float* __restrict__ w3,   // [2][128]
                             const float* __restrict__ b3,
                             float* __restrict__ jo) {
    __shared__ float s_sc[C_], s_sh[C_], s_w0[C_], s_w1[C_];
    int tid = threadIdx.x;
    if (tid < C_) {
        s_sc[tid] = scale[tid]; s_sh[tid] = shift[tid];
        s_w0[tid] = w3[tid];    s_w1[tid] = w3[C_ + tid];
    }
    __syncthreads();
    int p = blockIdx.x * 256 + tid;
    if (p >= NP_) return;
    float a0 = b3[0], a1 = b3[1];
    const v8h* yp = (const v8h*)(y2 + (size_t)p * C_);
#pragma unroll
    for (int cg = 0; cg < 16; ++cg) {
        v8h v = yp[cg];
#pragma unroll
        for (int i = 0; i < 8; ++i) {
            int ci = cg * 8 + i;
            float h = fmaxf(0.f, (float)v[i] * s_sc[ci] + s_sh[ci]);
            a0 += s_w0[ci] * h;
            a1 += s_w1[ci] * h;
        }
    }
    int b = p >> 14, sp = p & (HW_ - 1);
    jo[(b * 2 + 0) * HW_ + sp] = a0;
    jo[(b * 2 + 1) * HW_ + sp] = a1;
}

// ---------------------------------------------------------------------------
// Kernel 5: maxpool(2x2) of gj then convl (2->2, 3x3, pad 1) -> ds [B,2,64,64].
// ---------------------------------------------------------------------------
__global__ void poolconv_kernel(const float* __restrict__ outp,
                                const float* __restrict__ wl,
                                const float* __restrict__ bl,
                                float* __restrict__ dsb) {
    int idx = blockIdx.x * 256 + threadIdx.x;           // B*2*64*64
    if (idx >= B_ * 2 * 64 * 64) return;
    int xx = idx & 63, yy = (idx >> 6) & 63, co = (idx >> 12) & 1, b = idx >> 13;
    float acc = bl[co];
    for (int ci = 0; ci < 2; ++ci)
        for (int dy = 0; dy < 3; ++dy)
            for (int dx = 0; dx < 3; ++dx) {
                int py = yy + dy - 1, px = xx + dx - 1;
                float v = 0.f;
                if (py >= 0 && py < 64 && px >= 0 && px < 64) {
                    const float* g = outp + ((size_t)(b * 5 + ci) * H_ + py * 2) * W_ + px * 2;
                    v = fmaxf(fmaxf(g[0], g[1]), fmaxf(g[W_], g[W_ + 1]));
                }
                acc += wl[((co * 2 + ci) * 3 + dy) * 3 + dx] * v;
            }
    dsb[idx] = acc;
}

// ---------------------------------------------------------------------------
// Kernel 6/7: separable Gaussian splat of the keypoint mask.
// ---------------------------------------------------------------------------
__global__ void splat_h_kernel(const float* __restrict__ jt,
                               const float* __restrict__ alpha,
                               const float* __restrict__ beta,
                               float* __restrict__ sh) {
    int idx = blockIdx.x * 256 + threadIdx.x;
    if (idx >= NP_) return;
    float a = alpha[0] / (beta[0] * beta[0]);
    int x = idx & 127, y = (idx >> 7) & 127, b = idx >> 14;
    const float* col = jt + (size_t)b * HW_ + x;
    float acc = 0.f;
    for (int j = 0; j < H_; ++j) {
        if (col[j * W_] == 1.0f) {
            float d = (float)(y - j);
            acc += __expf(a * d * d);
        }
    }
    sh[idx] = acc;
}

__global__ void splat_w_kernel(const float* __restrict__ jt,
                               const float* __restrict__ sh,
                               const float* __restrict__ alpha,
                               const float* __restrict__ beta,
                               float* __restrict__ outg) {   // -> [1,B,H,W]
    int idx = blockIdx.x * 256 + threadIdx.x;
    if (idx >= NP_) return;
    float a = alpha[0] / (beta[0] * beta[0]);
    int x = idx & 127;
    int rowbase = idx & ~127;
    const float* srow = sh + rowbase;
    float acc = jt[idx];
    for (int i = 0; i < W_; ++i) {
        float d = (float)(x - i);
        acc += __expf(a * d * d) * srow[i];
    }
    float g = (acc <= 0.05f) ? 0.f : acc;
    g = (g > 1.f) ? 1.f : g;
    outg[idx] = g;
}

// ---------------------------------------------------------------------------
// Kernel 8: bilinear(64->128, align_corners) upsample of ds + jmap_origin,
// 2-class softmax -> sigmoid, plus lmap / joff heads.
// ---------------------------------------------------------------------------
__global__ void final_kernel(const float* __restrict__ outp,
                             const float* __restrict__ jo,
                             const float* __restrict__ dsb,
                             float* __restrict__ out) {
    int idx = blockIdx.x * 256 + threadIdx.x;
    if (idx >= NP_) return;
    int x = idx & 127, y = (idx >> 7) & 127, b = idx >> 14;
    int sp = idx & (HW_ - 1);

    float sy = y * (63.f / 127.f);
    int ly = (int)floorf(sy); ly = min(max(ly, 0), 62);
    float wy = sy - (float)ly;
    float sx = x * (63.f / 127.f);
    int lx = (int)floorf(sx); lx = min(max(lx, 0), 62);
    float wx = sx - (float)lx;

    float jm[2];
#pragma unroll
    for (int co = 0; co < 2; ++co) {
        const float* dd = dsb + (size_t)(b * 2 + co) * 64 * 64;
        float v00 = dd[ly * 64 + lx],       v01 = dd[ly * 64 + lx + 1];
        float v10 = dd[(ly + 1) * 64 + lx], v11 = dd[(ly + 1) * 64 + lx + 1];
        float up = (1.f - wy) * ((1.f - wx) * v00 + wx * v01)
                 +        wy  * ((1.f - wx) * v10 + wx * v11);
        jm[co] = jo[(b * 2 + co) * HW_ + sp] + up;
    }
    out[idx] = 1.f / (1.f + __expf(jm[0] - jm[1]));                       // jmap_pred
    out[NP_ + idx] = 1.f / (1.f + __expf(-outp[(size_t)(b * 5 + 2) * HW_ + sp])); // lmap
    float o3 = outp[(size_t)(b * 5 + 3) * HW_ + sp];
    float o4 = outp[(size_t)(b * 5 + 4) * HW_ + sp];
    out[2 * NP_ + (b * 2 + 0) * HW_ + sp] = 1.f / (1.f + __expf(-o3)) - 0.5f;
    out[2 * NP_ + (b * 2 + 1) * HW_ + sp] = 1.f / (1.f + __expf(-o4)) - 0.5f;
}

// ---------------------------------------------------------------------------
extern "C" void kernel_launch(void* const* d_in, const int* in_sizes, int n_in,
                              void* d_out, int out_size, void* d_ws, size_t ws_size,
                              hipStream_t stream) {
    (void)in_sizes; (void)n_in; (void)out_size; (void)ws_size;
    const float* outp = (const float*)d_in[0];
    const float* jt   = (const float*)d_in[1];
    const float* w1   = (const float*)d_in[2];
    const float* b1   = (const float*)d_in[3];
    const float* g1   = (const float*)d_in[4];
    const float* be1  = (const float*)d_in[5];
    const float* w2   = (const float*)d_in[6];
    const float* b2   = (const float*)d_in[7];
    const float* g2   = (const float*)d_in[8];
    const float* be2  = (const float*)d_in[9];
    const float* w3   = (const float*)d_in[10];
    const float* b3   = (const float*)d_in[11];
    const float* wl   = (const float*)d_in[12];
    const float* bl   = (const float*)d_in[13];
    const float* ga   = (const float*)d_in[14];
    const float* gb   = (const float*)d_in[15];

    char* ws = (char*)d_ws;
    _Float16* y1  = (_Float16*)(ws);                       // 134217728 B
    _Float16* y2  = (_Float16*)(ws + 134217728ull);        // 134217728 B
    _Float16* w1r = (_Float16*)(ws + 268435456ull);        // 8192 B
    _Float16* w2r = (_Float16*)(ws + 268443648ull);        // 294912 B
    float* sc1 = (float*)(ws + 268738560ull);              // 4x128 floats
    float* sh1 = sc1 + 128;
    float* sc2 = sc1 + 256;
    float* sh2 = sc1 + 384;
    float* jo  = (float*)(ws + 268740608ull);              // 4194304 B
    float* dsb = (float*)(ws + 272934912ull);              // 1048576 B
    float* shb = (float*)(ws + 273983488ull);              // 2097152 B
    float* ps1 = (float*)(ws + 276080640ull);              // SWG_*128 f32
    float* pq1 = ps1 + SWG_ * C_;
    float* ps2 = ps1 + 2 * SWG_ * C_;
    float* pq2 = ps1 + 3 * SWG_ * C_;                      // ends 277129216
    float* out = (float*)d_out;

    prep_weights_kernel<<<1024, 256, 0, stream>>>(w1, w2, w1r, w2r, ps1);
    conv1_kernel<<<32768, 256, 0, stream>>>(outp, w1r, b1, y1);
    bn_partial_kernel<<<SWG_, 256, 0, stream>>>(y1, ps1, pq1);
    bn_finalize_kernel<<<128, 256, 0, stream>>>(ps1, pq1, g1, be1, sc1, sh1);
    conv2_kernel<<<16384, 256, 0, stream>>>(y1, sc1, sh1, w2r, b2, y2);
    bn_partial_kernel<<<SWG_, 256, 0, stream>>>(y2, ps2, pq2);
    bn_finalize_kernel<<<128, 256, 0, stream>>>(ps2, pq2, g2, be2, sc2, sh2);
    conv3_kernel<<<2048, 256, 0, stream>>>(y2, sc2, sh2, w3, b3, jo);
    poolconv_kernel<<<1024, 256, 0, stream>>>(outp, wl, bl, dsb);
    splat_h_kernel<<<2048, 256, 0, stream>>>(jt, ga, gb, shb);
    splat_w_kernel<<<2048, 256, 0, stream>>>(jt, shb, ga, gb, out + 2097152);
    final_kernel<<<2048, 256, 0, stream>>>(outp, jo, dsb, out);
}
